// ThingGenerator_13821204758602
// MI455X (gfx1250) — matrix operand order repaired and checked by reading, not compiled, not run on hardware
//
#include <hip/hip_runtime.h>
#include <hip/hip_bf16.h>
#include <stdint.h>

// ---------------------------------------------------------------------------
// Problem constants (from reference): CIN=256, COUT=64, K=1024, H*W=80000
// ---------------------------------------------------------------------------
#define CIN   256
#define COUT  64
#define KNUM  1024
#define HW    80000
#define INST_SZ (KNUM * HW)          // 81,920,000 floats
#define SIM_THRES 0.9f

typedef __attribute__((ext_vector_type(16))) _Float16 v16h;
typedef __attribute__((ext_vector_type(8)))  _Float16 v8h;
typedef __attribute__((ext_vector_type(8)))  float    v8f;

union HFrag {
    v16h v;
    v8h  h[2];
    _Float16 e[16];
};

// A-matrix fragment (16x32 f16) from a row-major [rows][64] f16 matrix.
// Layout: lanes 0-15 -> halves 0-15 = K 0-15;
//         lanes16-31 -> halves 0-15 = K 16-31.
__device__ inline v16h load_a_frag(const _Float16* mat, int row, int kstep, int lane) {
    HFrag f;
    const _Float16* p = mat + row * 64 + kstep * 32;
    f.h[0] = *(const v8h*)(p + ((lane < 16) ? 0 : 16));
    f.h[1] = *(const v8h*)(p + ((lane < 16) ? 8 : 24));
    return f.v;
}

// B-matrix fragment (32x16 f16) where B[k][n] = mat[n_row][c0 + k], i.e. the
// B tile is rows of a row-major [rows][64] f16 matrix (used for a @ a^T).
// Layout: lane L holds n = L%16; halves 0-15 = K 0-15 (lanes<16) / 16-31.
__device__ inline v16h load_b_frag_rows(const _Float16* mat, int n_row, int kstep, int lane) {
    HFrag f;
    const _Float16* p = mat + n_row * 64 + kstep * 32 + ((lane < 16) ? 0 : 16);
    f.h[0] = *(const v8h*)(p);
    f.h[1] = *(const v8h*)(p + 8);
    return f.v;
}

// ---------------------------------------------------------------------------
// Stage 1: mw[k][c] = b[c] + sum_i feat[i][k] * W[c][i]     (1024 x 64, f32)
// feat = idx_feat viewed as [CIN][KNUM]
// ---------------------------------------------------------------------------
__global__ void k_mw(const float* __restrict__ feat, const float* __restrict__ W,
                     const float* __restrict__ b, float* __restrict__ mw) {
    int t = blockIdx.x * blockDim.x + threadIdx.x;   // 65536 threads
    int k = t & (KNUM - 1);
    int c = t >> 10;
    float acc = b[c];
    #pragma unroll 4
    for (int i = 0; i < CIN; ++i)
        acc = fmaf(feat[i * KNUM + k], W[c * CIN + i], acc);
    mw[k * COUT + c] = acc;
}

// ---------------------------------------------------------------------------
// Stage 2: a[k][:] = mw[k][:] / max(||mw[k]||, 1e-8)  -> f16
// ---------------------------------------------------------------------------
__global__ void k_norm(const float* __restrict__ mw, _Float16* __restrict__ a16) {
    int k = blockIdx.x * blockDim.x + threadIdx.x;   // 1024 threads
    float s = 0.f;
    #pragma unroll
    for (int c = 0; c < COUT; ++c) {
        float v = mw[k * COUT + c];
        s = fmaf(v, v, s);
    }
    float inv = 1.f / fmaxf(sqrtf(s), 1e-8f);
    #pragma unroll
    for (int c = 0; c < COUT; ++c)
        a16[k * COUT + c] = (_Float16)(mw[k * COUT + c] * inv);
}

// ---------------------------------------------------------------------------
// Stage 3: sim = a @ a^T via WMMA; label = triu(sim>=T) & same-class
// One wave computes one 16x16 tile (2 WMMA steps over c=64).
// Branchless epilogue: cate[] rows preloaded as int4 before the WMMAs.
// ---------------------------------------------------------------------------
__global__ void k_sim_label(const _Float16* __restrict__ a16,
                            const int* __restrict__ cate,
                            uint8_t* __restrict__ label) {
    int lane = threadIdx.x & 31;
    int wave = threadIdx.x >> 5;
    int tile = blockIdx.x * 8 + wave;                // 4096 tiles
    int i0 = (tile >> 6) * 16;
    int j0 = (tile & 63) * 16;

    int j   = j0 + (lane & 15);
    int ihi = i0 + ((lane >= 16) ? 8 : 0);

    // Issue the scalar-ish loads first so they overlap with WMMA latency.
    int cj  = cate[j];
    int4 ci0 = *(const int4*)(cate + ihi);           // ihi is 8-aligned -> 32B aligned
    int4 ci1 = *(const int4*)(cate + ihi + 4);

    v16h a0 = load_a_frag(a16, i0 + (lane & 15), 0, lane);
    v16h a1 = load_a_frag(a16, i0 + (lane & 15), 1, lane);
    v16h b0 = load_b_frag_rows(a16, j0 + (lane & 15), 0, lane);
    v16h b1 = load_b_frag_rows(a16, j0 + (lane & 15), 1, lane);

    v8f acc = {};
    acc = __builtin_amdgcn_wmma_f32_16x16x32_f16(false, a0, false, b0,
                                                 (short)0, acc, false, false);
    acc = __builtin_amdgcn_wmma_f32_16x16x32_f16(false, a1, false, b1,
                                                 (short)0, acc, false, false);

    int ci[8] = {ci0.x, ci0.y, ci0.z, ci0.w, ci1.x, ci1.y, ci1.z, ci1.w};
    #pragma unroll
    for (int r = 0; r < 8; ++r) {
        int i = ihi + r;
        // bitwise & keeps this branchless (v_cmp + v_cndmask, no execmask forks)
        uint8_t lab = (uint8_t)((acc[r] >= SIM_THRES) & (i <= j) & (ci[r] == cj));
        label[i * KNUM + j] = lab;
    }
}

// ---------------------------------------------------------------------------
// Stage 4: per-column cumulative scan. Thread j owns column j.
//   cum[i][j] = (sum_{r<=i} label[r][j]) < 2 ; keep[j] = cum[j][j]
//   lm0[i][j] = label[i][j] & cum[i][j]
// Row-order access -> fully coalesced across the 1024 threads.
// ---------------------------------------------------------------------------
__global__ void k_scan(const uint8_t* __restrict__ label,
                       uint8_t* __restrict__ lm0,
                       uint8_t* __restrict__ keep8,
                       float* __restrict__ keep_out) {
    int j = blockIdx.x * blockDim.x + threadIdx.x;   // 1024 threads
    int cnt = 0;
    uint8_t kp = 0;
    for (int i = 0; i < KNUM; ++i) {
        uint8_t v = label[i * KNUM + j];
        cnt += v;
        uint8_t cum = (cnt < 2) ? 1 : 0;
        if (i == j) kp = cum;
        lm0[i * KNUM + j] = v & cum;
    }
    keep8[j] = kp;
    keep_out[j] = (float)kp;
}

// ---------------------------------------------------------------------------
// Stage 5: merged[i][c] = keep[i] * (sum_j lm0[i][j]*mw[j][c]) / max(cnt,1)
// One block (64 threads) per row i; f32 accuracy, output stored as f16.
// ---------------------------------------------------------------------------
__global__ void k_merge(const uint8_t* __restrict__ lm0,
                        const uint8_t* __restrict__ keep8,
                        const float* __restrict__ mw,
                        _Float16* __restrict__ mg16) {
    int i = blockIdx.x;
    int c = threadIdx.x;
    float acc = 0.f;
    int cnt = 0;
    for (int j = 0; j < KNUM; ++j) {
        if (lm0[i * KNUM + j]) {
            acc += mw[j * COUT + c];
            ++cnt;
        }
    }
    float v = keep8[i] ? (acc / (float)(cnt > 0 ? cnt : 1)) : 0.f;
    mg16[i * COUT + c] = (_Float16)v;
}

// ---------------------------------------------------------------------------
// Stage 6: inst[k][p] = sum_c merged[k][c] * x[c][p]   via WMMA.
// Block = 8 waves; wave w owns p-strip [p0, p0+16). It loads its two B
// fragments (all 64 c of x, f32->f16 on the fly) ONCE into registers, then
// loops over all 64 M-tiles reusing them -> x is read exactly once from HBM.
// inst (327.7 MB > 192 MB L2) is written with non-temporal stores and x is
// read non-temporally: pure streaming, ~14 us floor at 23.3 TB/s.
// ---------------------------------------------------------------------------
__global__ void k_einsum(const float* __restrict__ x,
                         const _Float16* __restrict__ mg16,
                         float* __restrict__ inst) {
    int lane = threadIdx.x & 31;
    int wave = threadIdx.x >> 5;
    int p0 = blockIdx.x * 128 + wave * 16;           // 625 blocks * 128 = 80000
    int pcol = p0 + (lane & 15);

    HFrag bf0, bf1;
    {
        int cbase0 = (lane < 16) ? 0 : 16;
        #pragma unroll
        for (int hh = 0; hh < 16; ++hh) {
            bf0.e[hh] = (_Float16)__builtin_nontemporal_load(
                            &x[(cbase0 + hh) * HW + pcol]);
            bf1.e[hh] = (_Float16)__builtin_nontemporal_load(
                            &x[(cbase0 + 32 + hh) * HW + pcol]);
        }
    }

    int m = lane & 15;
    int rhi = (lane >= 16) ? 8 : 0;

    for (int mt = 0; mt < 64; ++mt) {
        int k0 = mt * 16;
        v16h a0 = load_a_frag(mg16, k0 + m, 0, lane);
        v16h a1 = load_a_frag(mg16, k0 + m, 1, lane);

        v8f acc = {};
        acc = __builtin_amdgcn_wmma_f32_16x16x32_f16(false, a0, false, bf0.v,
                                                     (short)0, acc, false, false);
        acc = __builtin_amdgcn_wmma_f32_16x16x32_f16(false, a1, false, bf1.v,
                                                     (short)0, acc, false, false);

        float* outp = inst + (size_t)(k0 + rhi) * HW + pcol;
        #pragma unroll
        for (int r = 0; r < 8; ++r)
            __builtin_nontemporal_store(acc[r], outp + (size_t)r * HW);
    }
}

// ---------------------------------------------------------------------------
// Stage 7: tail outputs: pred_cate (as float) and pred_score
// ---------------------------------------------------------------------------
__global__ void k_tail(const int* __restrict__ cate,
                       const float* __restrict__ score,
                       float* __restrict__ out) {
    int t = blockIdx.x * blockDim.x + threadIdx.x;   // 2048 threads
    if (t < KNUM)
        out[t] = (float)cate[t];
    else
        out[t] = score[t - KNUM];
}

// ---------------------------------------------------------------------------
extern "C" void kernel_launch(void* const* d_in, const int* in_sizes, int n_in,
                              void* d_out, int out_size, void* d_ws, size_t ws_size,
                              hipStream_t stream) {
    const float* x     = (const float*)d_in[0];   // [64][80000]
    const float* feat  = (const float*)d_in[1];   // [256][1024] (idx_feat)
    const int*   cate  = (const int*)d_in[2];     // [1024]
    const float* score = (const float*)d_in[3];   // [1024]
    const float* W     = (const float*)d_in[4];   // [64][256]
    const float* b     = (const float*)d_in[5];   // [64]

    float* out = (float*)d_out;

    char* ws = (char*)d_ws;
    float*    mw    = (float*)(ws + 0);            // 256 KB
    _Float16* a16   = (_Float16*)(ws + 262144);    // 128 KB
    uint8_t*  label = (uint8_t*)(ws + 393216);     // 1 MB
    uint8_t*  lm0   = (uint8_t*)(ws + 1441792);    // 1 MB
    uint8_t*  keep8 = (uint8_t*)(ws + 2490368);    // 1 KB
    _Float16* mg16  = (_Float16*)(ws + 2491392);   // 128 KB  (total ~2.5 MB)

    // 1) mw = feat^T W^T + b
    k_mw<<<256, 256, 0, stream>>>(feat, W, b, mw);
    // 2) a = normalize(mw) -> f16
    k_norm<<<4, 256, 0, stream>>>(mw, a16);
    // 3) sim = a a^T (WMMA) -> label
    k_sim_label<<<512, 256, 0, stream>>>(a16, cate, label);
    // 4) column scans -> lm0, keep
    k_scan<<<4, 256, 0, stream>>>(label, lm0, keep8, out + INST_SZ);
    // 5) merged rows (f32 accumulate) -> f16
    k_merge<<<1024, 64, 0, stream>>>(lm0, keep8, mw, mg16);
    // 6) inst = merged @ x (WMMA, store-bound ~14us at 23.3 TB/s)
    k_einsum<<<625, 256, 0, stream>>>(x, mg16, out);
    // 7) pred_cate / pred_score passthrough
    k_tail<<<8, 256, 0, stream>>>(cate, score, out + INST_SZ + KNUM);
}
